// RegimeDetectionModule_53102975647886
// MI455X (gfx1250) — matrix operand order repaired and checked by reading, not compiled, not run on hardware
//
#include <hip/hip_runtime.h>
#include <hip/hip_bf16.h>
#include <math.h>

// RegimeDetectionModule fused kernels for MI455X (gfx1250, wave32, WMMA).
// B=32 S=2048 D=512 R=4 H=32.  Tokens T = 65536.
// GEMM chain in f16 WMMA (f32 accumulate); GRU sequential, one wave/batch.

typedef __attribute__((ext_vector_type(16))) _Float16     v16h;
typedef __attribute__((ext_vector_type(8)))  float        v8f;
typedef __attribute__((ext_vector_type(4)))  unsigned int u32x4;

// explicit address-space pointers: force global_load_b128 / ds_load_b128
typedef const __attribute__((address_space(1))) u32x4* gu4p;
typedef const __attribute__((address_space(3))) u32x4* lu4p;

#define SS   2048
#define DD   512
#define HH   32
#define NTOK 65536
#define WPB  4                  // waves per block in the fused kernel
#define TPW  2                  // 16-token M-tiles per wave
// per-tile LDS slice: buf1 f16[16*384] (12288B) | buf2 f32[16*64] (4096B)
//                     slog f32[16*4] (256B)     | sprob f32[16*4] (256B)
#define SLICE     16896
#define LDS_TOTAL (SLICE * TPW * WPB)   // 135168 B < 320KB/WGP

// ---- fragment load helpers -------------------------------------------------
// B fragment: 16 contiguous halfs (32B) of one weight row (global).
__device__ __forceinline__ v16h ld_b32h_g(const _Float16* p) {
  gu4p q = (gu4p)(const u32x4*)p;
  union { u32x4 u[2]; v16h h; } t;
  t.u[0] = q[0];
  t.u[1] = q[1];
  return t.h;
}
// A fragment from global: chunks at k1..k1+7 and k2..k2+7 of one token row.
__device__ __forceinline__ v16h ld_a2_g(const _Float16* p, int k1, int k2) {
  union { u32x4 u[2]; v16h h; } t;
  t.u[0] = *(gu4p)(const u32x4*)(p + k1);
  t.u[1] = *(gu4p)(const u32x4*)(p + k2);
  return t.h;
}
// A fragment from LDS.
__device__ __forceinline__ v16h ld_a2_l(const _Float16* p, int k1, int k2) {
  union { u32x4 u[2]; v16h h; } t;
  t.u[0] = *(lu4p)(const u32x4*)(p + k1);
  t.u[1] = *(lu4p)(const u32x4*)(p + k2);
  return t.h;
}
__device__ __forceinline__ v8f wmma16(v16h a, v16h b, v8f c) {
  return __builtin_amdgcn_wmma_f32_16x16x32_f16(false, a, false, b, (short)0, c,
                                                false, false);
}

// ---- prep: f32 -> f16 conversions, concatenated L1 weights -----------------
// wcat rows 0..127 = rw1, rows 128..383 = ew1 (flat [256][512]).
// bcat[0..127] = rb1, bcat[128..383] = eb1 (flat).
__global__ void cvt_weights(const float* __restrict__ rw1, const float* __restrict__ rb1,
                            const float* __restrict__ rw2,
                            const float* __restrict__ ew1, const float* __restrict__ eb1,
                            const float* __restrict__ ew2,
                            _Float16* __restrict__ wcat, _Float16* __restrict__ w2,
                            _Float16* __restrict__ e2, float* __restrict__ bcat) {
  int i = blockIdx.x * blockDim.x + threadIdx.x;
  if (i < 128 * 512)    wcat[i] = (_Float16)rw1[i];
  if (i < 4 * 64 * 512) wcat[128 * 512 + i] = (_Float16)ew1[i];
  if (i < 64 * 128)     w2[i] = (_Float16)rw2[i];
  if (i < 4 * 64 * 64)  e2[i] = (_Float16)ew2[i];
  if (i < 128)          bcat[i] = rb1[i];
  if (i < 256)          bcat[128 + i] = eb1[i];
}

__global__ void cvt_x(const float* __restrict__ x, _Float16* __restrict__ xf) {
  size_t i = ((size_t)blockIdx.x * blockDim.x + threadIdx.x) * 8;
  float4 a = *(const float4*)(x + i);
  float4 b = *(const float4*)(x + i + 4);
  union { _Float16 h[8]; uint4 u; } t;
  t.h[0] = (_Float16)a.x; t.h[1] = (_Float16)a.y;
  t.h[2] = (_Float16)a.z; t.h[3] = (_Float16)a.w;
  t.h[4] = (_Float16)b.x; t.h[5] = (_Float16)b.y;
  t.h[6] = (_Float16)b.z; t.h[7] = (_Float16)b.w;
  *(uint4*)(xf + i) = t.u;
}

// ---- fused router + experts + softmax + weighted + uncertainty -------------
__global__ __launch_bounds__(32 * WPB, 1)
void fused_moe_kernel(const _Float16* __restrict__ xf,
                      const _Float16* __restrict__ wcat, const float* __restrict__ bcat,
                      const _Float16* __restrict__ w2f,  const float* __restrict__ rb2,
                      const float* __restrict__ rw3,     const float* __restrict__ rb3,
                      const _Float16* __restrict__ e2w,  const float* __restrict__ eb2,
                      float* __restrict__ out_probs, float* __restrict__ out_w,
                      float* __restrict__ out_unc) {
  extern __shared__ char smem[];
  const int lane = threadIdx.x & 31;
  const int wid  = threadIdx.x >> 5;
  char* wavebase = smem + wid * (SLICE * TPW);

  const int tile0 = (blockIdx.x * WPB + wid) * TPW;   // two consecutive M-tiles
  const int row0  = tile0 * 16;                       // 32 tokens
  const int m     = lane & 15;                        // token row / weight row
  const int hi    = lane >> 4;
  const _Float16* xrow0 = xf + (size_t)(row0 + m) * DD;
  const _Float16* xrow1 = xf + (size_t)(row0 + 16 + m) * DD;

  _Float16* b1a = (_Float16*)wavebase;             // tile0: [16][384] halfs
  _Float16* b1b = (_Float16*)(wavebase + SLICE);   // tile1

  // ---------- Phase 1: h1 = relu(x@rw1^T+b) [N cols 0..127],
  //                     e1 = relu(x@ew1^T+b) [N cols 128..383]
  // 24 N-tiles of 16, K=512 (16 k-steps), 4 N-tiles x 2 M-tiles per pass.
  for (int g = 0; g < 6; ++g) {
    v8f acc0[4] = {}, acc1[4] = {};
#pragma unroll 2
    for (int kt = 0; kt < 16; ++kt) {
      const int k1 = kt * 32 + hi * 16;
      v16h a0 = ld_a2_g(xrow0, k1, k1 + 8);
      v16h a1 = ld_a2_g(xrow1, k1, k1 + 8);
      const int kb = kt * 32 + hi * 16;
#pragma unroll
      for (int i = 0; i < 4; ++i) {
        // weight row (g*4+i)*16+m of concatenated [384][512]: pure arithmetic
        v16h b = ld_b32h_g(wcat + (size_t)((g * 4 + i) * 16 + m) * DD + kb);
        acc0[i] = wmma16(a0, b, acc0[i]);
        acc1[i] = wmma16(a1, b, acc1[i]);
      }
    }
#pragma unroll
    for (int i = 0; i < 4; ++i) {
      const int col = (g * 4 + i) * 16 + m;
      const float bv = bcat[col];
#pragma unroll
      for (int v = 0; v < 8; ++v) {
        float v0 = fmaxf(acc0[i][v] + bv, 0.f);
        float v1 = fmaxf(acc1[i][v] + bv, 0.f);
        b1a[(v + 8 * hi) * 384 + col] = (_Float16)v0;
        b1b[(v + 8 * hi) * 384 + col] = (_Float16)v1;
      }
    }
  }
  __syncthreads();

  // ---------- Phases 2 & 3, per M-tile slice ----------
  for (int t2 = 0; t2 < TPW; ++t2) {
    char*     tb    = wavebase + t2 * SLICE;
    _Float16* buf1  = (_Float16*)tb;                    // [16][384] halfs
    float*    buf2  = (float*)(tb + 12288);             // [16][64]
    float*    slog  = (float*)(tb + 12288 + 4096);      // [16][4]
    float*    sprob = (float*)(tb + 12288 + 4096 + 256);
    const int rowt  = row0 + t2 * 16;

    // Phase 2a: h2 = relu(h1 @ rw2^T + b)   (K=128, N=64)
    {
      v8f hacc[4] = {};
#pragma unroll
      for (int kt = 0; kt < 4; ++kt) {
        const int k1 = kt * 32 + hi * 16;
        v16h a = ld_a2_l(buf1 + m * 384, k1, k1 + 8);
        const int kb = kt * 32 + hi * 16;
#pragma unroll
        for (int i = 0; i < 4; ++i) {
          v16h b = ld_b32h_g(w2f + (size_t)(i * 16 + m) * 128 + kb);
          hacc[i] = wmma16(a, b, hacc[i]);
        }
      }
#pragma unroll
      for (int i = 0; i < 4; ++i) {
        float bv = rb2[i * 16 + m];
#pragma unroll
        for (int v = 0; v < 8; ++v)
          buf2[(v + 8 * hi) * 64 + i * 16 + m] = fmaxf(hacc[i][v] + bv, 0.f);
      }
    }

    // Phase 2b: e2_r = relu(e1_r @ ew2_r^T + b); overwrite e1 region with f16
#pragma unroll
    for (int r = 0; r < 4; ++r) {
      v8f ec[4] = {};
#pragma unroll
      for (int kt = 0; kt < 2; ++kt) {
        const int k1 = kt * 32 + hi * 16;
        v16h a = ld_a2_l(buf1 + m * 384 + 128 + r * 64, k1, k1 + 8);
        const int kb = kt * 32 + hi * 16;
#pragma unroll
        for (int i = 0; i < 4; ++i) {
          v16h b = ld_b32h_g(e2w + (size_t)(r * 64 + i * 16 + m) * 64 + kb);
          ec[i] = wmma16(a, b, ec[i]);
        }
      }
#pragma unroll
      for (int i = 0; i < 4; ++i) {
        float bv = eb2[r * 64 + i * 16 + m];
#pragma unroll
        for (int v = 0; v < 8; ++v) {
          float val = fmaxf(ec[i][v] + bv, 0.f);
          buf1[(v + 8 * hi) * 384 + 128 + r * 64 + i * 16 + m] = (_Float16)val;
        }
      }
    }
    __syncthreads();

    // Phase 3: logits (lane -> row m, regimes hi*2, hi*2+1)
#pragma unroll
    for (int rr = 0; rr < 2; ++rr) {
      int r = hi * 2 + rr;
      float acc = rb3[r];
      const float* hrow = buf2 + m * 64;
      const float* wrow = rw3 + r * 64;
#pragma unroll 8
      for (int k = 0; k < 64; ++k) acc += hrow[k] * wrow[k];
      slog[m * 4 + r] = acc;
    }
    __syncthreads();

    if (lane < 16) {
      float l0 = slog[lane * 4 + 0], l1 = slog[lane * 4 + 1];
      float l2 = slog[lane * 4 + 2], l3 = slog[lane * 4 + 3];
      float mx = fmaxf(fmaxf(l0, l1), fmaxf(l2, l3));
      float e0 = __expf(l0 - mx), e1v = __expf(l1 - mx);
      float e2v = __expf(l2 - mx), e3 = __expf(l3 - mx);
      float inv = 1.f / (e0 + e1v + e2v + e3);
      float p0 = e0 * inv, p1 = e1v * inv, p2 = e2v * inv, p3 = e3 * inv;
      sprob[lane * 4 + 0] = p0; sprob[lane * 4 + 1] = p1;
      sprob[lane * 4 + 2] = p2; sprob[lane * 4 + 3] = p3;
      float4 pv; pv.x = p0; pv.y = p1; pv.z = p2; pv.w = p3;
      *(float4*)(out_probs + (size_t)(rowt + lane) * 4) = pv;
      float mean = 0.25f * (p0 + p1 + p2 + p3);
      float d0 = p0 - mean, d1 = p1 - mean, d2 = p2 - mean, d3 = p3 - mean;
      out_unc[rowt + lane] =
          sqrtf((d0 * d0 + d1 * d1 + d2 * d2 + d3 * d3) * (1.f / 3.f));
    }
    __syncthreads();

    // weighted[m][c] = sum_r probs[m][r] * e2[m][r][c]; lane -> (m, 32 cols)
    {
      const int cb = hi * 32;
      float p0 = sprob[m * 4 + 0], p1 = sprob[m * 4 + 1];
      float p2 = sprob[m * 4 + 2], p3 = sprob[m * 4 + 3];
      const _Float16* er = buf1 + m * 384 + 128;
      float* orow = out_w + (size_t)(rowt + m) * 64 + cb;
#pragma unroll
      for (int c4 = 0; c4 < 8; ++c4) {
        float4 o;
#pragma unroll
        for (int q = 0; q < 4; ++q) {
          int c = cb + c4 * 4 + q;
          float w = p0 * (float)er[c]       + p1 * (float)er[64 + c]
                  + p2 * (float)er[128 + c] + p3 * (float)er[192 + c];
          ((float*)&o)[q] = w;
        }
        *(float4*)(orow + c4 * 4) = o;
      }
    }
    __syncthreads();
  }
}

// ---- GRU over probs: one wave32 per batch, lane j = hidden unit j ----------
__global__ __launch_bounds__(32)
void gru_kernel(const float* __restrict__ probs,
                const float* __restrict__ wih, const float* __restrict__ whh,
                const float* __restrict__ bih, const float* __restrict__ bhh,
                float* __restrict__ trans) {
  const int b = blockIdx.x;
  const int j = threadIdx.x;           // 0..31
  float Wi[3][4], Wh[3][32], Bi[3], Bh[3];
#pragma unroll
  for (int g = 0; g < 3; ++g) {
    int row = g * HH + j;
#pragma unroll
    for (int k = 0; k < 4; ++k)  Wi[g][k] = wih[row * 4 + k];
#pragma unroll
    for (int k = 0; k < 32; ++k) Wh[g][k] = whh[row * 32 + k];
    Bi[g] = bih[row]; Bh[g] = bhh[row];
  }
  float h = 0.f;
  const float* pb = probs + (size_t)b * SS * 4;
  float* tb = trans + (size_t)b * SS * HH;
  for (int t = 0; t < SS; ++t) {
    float4 p = *(const float4*)(pb + (size_t)t * 4);
    float gr = Bi[0] + Wi[0][0]*p.x + Wi[0][1]*p.y + Wi[0][2]*p.z + Wi[0][3]*p.w;
    float gz = Bi[1] + Wi[1][0]*p.x + Wi[1][1]*p.y + Wi[1][2]*p.z + Wi[1][3]*p.w;
    float gn = Bi[2] + Wi[2][0]*p.x + Wi[2][1]*p.y + Wi[2][2]*p.z + Wi[2][3]*p.w;
    float hr = Bh[0], hz = Bh[1], hn = Bh[2];
#pragma unroll
    for (int k = 0; k < 32; ++k) {
      float hk = __shfl(h, k, 32);     // broadcast h[k] across the wave
      hr += Wh[0][k] * hk;
      hz += Wh[1][k] * hk;
      hn += Wh[2][k] * hk;
    }
    float r = 1.f / (1.f + __expf(-(gr + hr)));
    float z = 1.f / (1.f + __expf(-(gz + hz)));
    float n = tanhf(gn + r * hn);
    h = (1.f - z) * n + z * h;
    tb[(size_t)t * HH + j] = h;
  }
}

// ---- launcher --------------------------------------------------------------
extern "C" void kernel_launch(void* const* d_in, const int* in_sizes, int n_in,
                              void* d_out, int out_size, void* d_ws, size_t ws_size,
                              hipStream_t stream) {
  const float* x     = (const float*)d_in[0];
  const float* rw1   = (const float*)d_in[1];
  const float* rb1   = (const float*)d_in[2];
  const float* rw2   = (const float*)d_in[3];
  const float* rb2   = (const float*)d_in[4];
  const float* rw3   = (const float*)d_in[5];
  const float* rb3   = (const float*)d_in[6];
  const float* ew1   = (const float*)d_in[7];
  const float* eb1   = (const float*)d_in[8];
  const float* ew2   = (const float*)d_in[9];
  const float* eb2   = (const float*)d_in[10];
  const float* gw_ih = (const float*)d_in[11];
  const float* gw_hh = (const float*)d_in[12];
  const float* gb_ih = (const float*)d_in[13];
  const float* gb_hh = (const float*)d_in[14];

  float* out       = (float*)d_out;
  float* out_probs = out;                                  // [32,2048,4]
  float* out_w     = out + 262144;                         // [32,2048,64]
  float* out_tr    = out + 262144 + 4194304;               // [32,2048,32]
  float* out_unc   = out + 262144 + 4194304 + 2097152;     // [32,2048]

  // workspace layout (bytes), 16B-aligned segments, total ~64.5 MB
  char* ws = (char*)d_ws;
  _Float16* wcat = (_Float16*)(ws);                // [384][512] halfs (rw1;ew1)
  _Float16* w2f  = (_Float16*)(ws + 393216);       // [64][128]
  _Float16* e2f  = (_Float16*)(ws + 409600);       // [256][64]
  float*    bcat = (float*)(ws + 442368);          // [384] f32 (rb1;eb1)
  _Float16* xf   = (_Float16*)(ws + 443904);       // [65536][512] halfs

  cvt_weights<<<512, 256, 0, stream>>>(rw1, rb1, rw2, ew1, eb1, ew2,
                                       wcat, w2f, e2f, bcat);
  cvt_x<<<16384, 256, 0, stream>>>(x, xf);
  fused_moe_kernel<<<NTOK / (16 * TPW * WPB), 32 * WPB, LDS_TOTAL, stream>>>(
      xf, wcat, bcat, w2f, rb2, rw3, rb3, e2f, eb2,
      out_probs, out_w, out_unc);
  gru_kernel<<<32, 32, 0, stream>>>(out_probs, gw_ih, gw_hh, gb_ih, gb_hh, out_tr);
}
